// Vim_29248727285916
// MI455X (gfx1250) — matrix operand order repaired and checked by reading, not compiled, not run on hardware
//
#include <hip/hip_runtime.h>
#include <hip/hip_bf16.h>

// ---------------------------------------------------------------------------
// Vision Mamba (Vim) forward: patch-embed -> 12 x bidirectional SSM block ->
// mean-pool -> LN -> linear -> softmax.   All dense projections run on the
// CDNA5 matrix cores via v_wmma_f32_16x16x32_f16 (f32 accumulate); each wave
// drives two independent accumulator chains off one shared A fragment to hide
// XDL latency.  The sequential selective scan runs one wave32 per
// (batch, channel) with the 64-wide state held in 2 VGPRs/lane.
// ---------------------------------------------------------------------------

#define DIMC   256
#define SEQ    196
#define BATCH  16
#define MTOK   (BATCH * SEQ)      // 3136 tokens
#define DSTATE 64
#define DTRANK 32
#define EDBC   (DTRANK + 2 * DSTATE)  // 160
#define DEPTHL 12
#define NCLS   1000

typedef __attribute__((ext_vector_type(16))) _Float16 v16h;
typedef __attribute__((ext_vector_type(8)))  float    v8f;
typedef __attribute__((ext_vector_type(4)))  float    f32x4;
typedef __attribute__((ext_vector_type(2)))  float    f32x2;

union FragU {
    struct { f32x4 lo, hi; } q;
    v16h v;
};

// A-fragment: halves 0..7 at p[0..7], halves 8..15 at p[16..23]
__device__ __forceinline__ v16h ld_frag_a(const _Float16* p) {
    FragU u;
    u.q.lo = *reinterpret_cast<const f32x4*>(p);
    u.q.hi = *reinterpret_cast<const f32x4*>(p + 16);
    return u.v;
}
// B-fragment: halves 0..7 at p[0..7], halves 8..15 at p[16..23]
__device__ __forceinline__ v16h ld_frag_b(const _Float16* p) {
    FragU u;
    u.q.lo = *reinterpret_cast<const f32x4*>(p);
    u.q.hi = *reinterpret_cast<const f32x4*>(p + 16);
    return u.v;
}

__device__ __forceinline__ float softplusf(float x) {
    return x > 20.f ? x : log1pf(__expf(x));
}

// Single-tile GEMM: A (16 x K) in LDS row-major f16, B = weight row n_g (f16).
__device__ __forceinline__ v8f gemm_acc(const _Float16* __restrict__ aSmem, int lda,
                                        const _Float16* __restrict__ wrow, int kblocks,
                                        v8f acc, int lane) {
    const int m    = lane & 15;
    const int aoff = (lane & 16) ? 8 : 0;
    const int boff = (lane & 16) ? 8 : 0;
    const _Float16* ap = aSmem + m * lda + aoff;
    const _Float16* bp = wrow + boff;
#pragma unroll 4
    for (int kb = 0; kb < kblocks; ++kb) {
        v16h a = ld_frag_a(ap); ap += 32;
        v16h b = ld_frag_b(bp); bp += 32;
        acc = __builtin_amdgcn_wmma_f32_16x16x32_f16(
            false, a, false, b, (short)0, acc, false, false);
    }
    return acc;
}

// Dual-tile GEMM: one shared A fragment feeds two independent WMMA chains.
__device__ __forceinline__ void gemm2_acc(const _Float16* __restrict__ aSmem, int lda,
                                          const _Float16* __restrict__ w0,
                                          const _Float16* __restrict__ w1,
                                          int kblocks, v8f& acc0, v8f& acc1, int lane) {
    const int m    = lane & 15;
    const int aoff = (lane & 16) ? 8 : 0;
    const int boff = (lane & 16) ? 8 : 0;
    const _Float16* ap  = aSmem + m * lda + aoff;
    const _Float16* bp0 = w0 + boff;
    const _Float16* bp1 = w1 + boff;
#pragma unroll 4
    for (int kb = 0; kb < kblocks; ++kb) {
        v16h a  = ld_frag_a(ap);  ap  += 32;
        v16h b0 = ld_frag_b(bp0); bp0 += 32;
        v16h b1 = ld_frag_b(bp1); bp1 += 32;
        acc0 = __builtin_amdgcn_wmma_f32_16x16x32_f16(
            false, a, false, b0, (short)0, acc0, false, false);
        acc1 = __builtin_amdgcn_wmma_f32_16x16x32_f16(
            false, a, false, b1, (short)0, acc1, false, false);
    }
}

// ---------------------------------------------------------------------------
__global__ void k_f32_to_f16(const float* __restrict__ s, _Float16* __restrict__ d, int n) {
    int i = blockIdx.x * 256 + threadIdx.x;
    if (i < n) d[i] = (_Float16)s[i];
}

// ---------------------------------------------------------------------------
// Patch embed: t = xp @ patch_w^T + patch_b.  xp gathered from NCHW image.
__global__ void k_patch(const float* __restrict__ xin,
                        const _Float16* __restrict__ pw16,
                        const float* __restrict__ pb,
                        float* __restrict__ tout) {
    __shared__ __attribute__((aligned(16))) _Float16 xp16[16 * 768];
    const int tid = threadIdx.x, wv = tid >> 5, lane = tid & 31;
    const int tok0 = blockIdx.x * 16;
    for (int e = tid; e < 16 * 768; e += 256) {
        int t = e / 768, p = e - t * 768;
        int g = tok0 + t;
        int b = g / SEQ, s = g - b * SEQ;
        int hp = s / 14, wp = s - hp * 14;
        int c = p % 3, q = p / 3;
        int p2 = q & 15, p1 = q >> 4;
        size_t src = (((size_t)b * 3 + c) * 224 + (hp * 16 + p1)) * 224 + (wp * 16 + p2);
        xp16[e] = (_Float16)xin[src];
    }
    __syncthreads();
    const int moff = (lane & 16) ? 8 : 0;
    const int c0 = (wv << 4) + (lane & 15);
    const int c1 = c0 + 128;
    v8f a0, a1;
    for (int r = 0; r < 8; ++r) { a0[r] = pb[c0]; a1[r] = pb[c1]; }
    gemm2_acc(xp16, 768, pw16 + (size_t)c0 * 768, pw16 + (size_t)c1 * 768, 24, a0, a1, lane);
    for (int r = 0; r < 8; ++r) {
        size_t rg = (size_t)(tok0 + r + moff) * DIMC;
        tout[rg + c0] = a0[r];
        tout[rg + c1] = a1[r];
    }
}

// ---------------------------------------------------------------------------
// LayerNorm + z = xn @ pw^T + pb.  Stores z (f32) and z (f16, A-staging).
__global__ void k_ln_proj(const float* __restrict__ xin,
                          const float* __restrict__ ng, const float* __restrict__ nb,
                          const _Float16* __restrict__ pw16, const float* __restrict__ pb,
                          float* __restrict__ zout, _Float16* __restrict__ z16g) {
    __shared__ __attribute__((aligned(16))) float    xr[16 * DIMC];
    __shared__ __attribute__((aligned(16))) _Float16 xnt[16 * DIMC];
    const int tid = threadIdx.x, wv = tid >> 5, lane = tid & 31;
    const int tok0 = blockIdx.x * 16;
    {
        const f32x4* g = (const f32x4*)(xin + (size_t)tok0 * DIMC);
        f32x4* d = (f32x4*)xr;
        for (int i = 0; i < 4; ++i) d[tid + i * 256] = g[tid + i * 256];
    }
    __syncthreads();
    // per-row LN stats: wave w handles rows 2w, 2w+1 in its two 16-lane halves
    const int row = (wv << 1) | (lane >> 4);
    const int sl = lane & 15;
    const float* xrow = xr + row * DIMC;
    float sum = 0.f, sq = 0.f;
#pragma unroll
    for (int i = 0; i < 16; ++i) { float v = xrow[sl + (i << 4)]; sum += v; sq += v * v; }
    for (int msk = 8; msk >= 1; msk >>= 1) {
        sum += __shfl_xor(sum, msk, 32);
        sq  += __shfl_xor(sq,  msk, 32);
    }
    const float mean = sum * (1.f / DIMC);
    const float var  = sq * (1.f / DIMC) - mean * mean;
    const float rstd = rsqrtf(var + 1e-5f);
#pragma unroll
    for (int i = 0; i < 16; ++i) {
        int e = sl + (i << 4);
        xnt[row * DIMC + e] = (_Float16)((xrow[e] - mean) * rstd * ng[e] + nb[e]);
    }
    __syncthreads();
    const int moff = (lane & 16) ? 8 : 0;
    const int c0 = (wv << 4) + (lane & 15);
    const int c1 = c0 + 128;
    v8f a0, a1;
    for (int r = 0; r < 8; ++r) { a0[r] = pb[c0]; a1[r] = pb[c1]; }
    gemm2_acc(xnt, DIMC, pw16 + (size_t)c0 * DIMC, pw16 + (size_t)c1 * DIMC, 8, a0, a1, lane);
    for (int r = 0; r < 8; ++r) {
        size_t rg = (size_t)(tok0 + r + moff) * DIMC;
        zout[rg + c0] = a0[r]; z16g[rg + c0] = (_Float16)a0[r];
        zout[rg + c1] = a1[r]; z16g[rg + c1] = (_Float16)a1[r];
    }
}

// ---------------------------------------------------------------------------
// dbc epilogue: delta_raw -> LDS f16; Bm/Cm -> global f32 with (n, n+32)
// interleaved so the scan can load each state pair as one b64.
__device__ __forceinline__ void dbc_store(v8f acc, int col, int tok0, int moff,
                                          _Float16* __restrict__ draw,
                                          float* __restrict__ BmO, float* __restrict__ CmO) {
#pragma unroll
    for (int r = 0; r < 8; ++r) {
        int row = r + moff;
        float v = acc[r];
        if (col < DTRANK) {
            draw[row * DTRANK + col] = (_Float16)v;
        } else if (col < DTRANK + DSTATE) {
            int n = col - DTRANK;
            BmO[(size_t)(tok0 + row) * DSTATE + ((n & 31) << 1) + (n >> 5)] = v;
        } else {
            int n = col - DTRANK - DSTATE;
            CmO[(size_t)(tok0 + row) * DSTATE + ((n & 31) << 1) + (n >> 5)] = v;
        }
    }
}

// One direction: xc = softplus(z@cw^T+cb); dbc = xc@dbc_w^T;
// delta = softplus(dbc[:,:32]@dt_w^T + dt_b); Bm = dbc[:,32:96]; Cm = dbc[:,96:160]
__global__ void k_conv_ssm(const _Float16* __restrict__ z16g,
                           const _Float16* __restrict__ cw16, const float* __restrict__ cb,
                           const _Float16* __restrict__ dbcw16,
                           const _Float16* __restrict__ dtw16, const float* __restrict__ dtb,
                           float* __restrict__ xcout, float* __restrict__ BmO,
                           float* __restrict__ CmO, float* __restrict__ deltaO) {
    __shared__ __attribute__((aligned(16))) _Float16 zt[16 * DIMC];
    __shared__ __attribute__((aligned(16))) _Float16 xct[16 * DIMC];
    __shared__ __attribute__((aligned(16))) _Float16 draw[16 * DTRANK];
    const int tid = threadIdx.x, wv = tid >> 5, lane = tid & 31;
    const int tok0 = blockIdx.x * 16;
    {   // stage z tile (f16) from global: 8KB as b128s
        const f32x4* g = (const f32x4*)(z16g + (size_t)tok0 * DIMC);
        f32x4* d = (f32x4*)zt;
        for (int i = 0; i < 2; ++i) d[tid + i * 256] = g[tid + i * 256];
    }
    __syncthreads();
    const int moff = (lane & 16) ? 8 : 0;
    const int l15 = lane & 15;
    // GEMM1: xc = softplus(z @ cw^T + cb)   (dual-tile per wave)
    {
        const int c0 = (wv << 4) + l15, c1 = c0 + 128;
        v8f a0, a1;
        for (int r = 0; r < 8; ++r) { a0[r] = cb[c0]; a1[r] = cb[c1]; }
        gemm2_acc(zt, DIMC, cw16 + (size_t)c0 * DIMC, cw16 + (size_t)c1 * DIMC, 8, a0, a1, lane);
        for (int r = 0; r < 8; ++r) {
            int row = r + moff;
            float v0 = softplusf(a0[r]);
            float v1 = softplusf(a1[r]);
            xcout[(size_t)(tok0 + row) * DIMC + c0] = v0;
            xcout[(size_t)(tok0 + row) * DIMC + c1] = v1;
            xct[row * DIMC + c0] = (_Float16)v0;
            xct[row * DIMC + c1] = (_Float16)v1;
        }
    }
    __syncthreads();
    // GEMM2: dbc (160 cols): waves 0,1 dual (tiles w,w+8); waves 2..7 single
    if (wv < 2) {
        const int c0 = (wv << 4) + l15, c1 = c0 + 128;
        v8f a0, a1;
        for (int r = 0; r < 8; ++r) { a0[r] = 0.f; a1[r] = 0.f; }
        gemm2_acc(xct, DIMC, dbcw16 + (size_t)c0 * DIMC, dbcw16 + (size_t)c1 * DIMC, 8, a0, a1, lane);
        dbc_store(a0, c0, tok0, moff, draw, BmO, CmO);
        dbc_store(a1, c1, tok0, moff, draw, BmO, CmO);
    } else {
        const int c0 = (wv << 4) + l15;
        v8f a0;
        for (int r = 0; r < 8; ++r) a0[r] = 0.f;
        a0 = gemm_acc(xct, DIMC, dbcw16 + (size_t)c0 * DIMC, 8, a0, lane);
        dbc_store(a0, c0, tok0, moff, draw, BmO, CmO);
    }
    __syncthreads();
    // GEMM3: delta = softplus(delta_raw @ dt_w^T + dt_b), K = 32 (dual-tile)
    {
        const int c0 = (wv << 4) + l15, c1 = c0 + 128;
        v8f a0, a1;
        for (int r = 0; r < 8; ++r) { a0[r] = dtb[c0]; a1[r] = dtb[c1]; }
        gemm2_acc(draw, DTRANK, dtw16 + (size_t)c0 * DTRANK, dtw16 + (size_t)c1 * DTRANK, 1, a0, a1, lane);
        for (int r = 0; r < 8; ++r) {
            size_t rg = (size_t)(tok0 + r + moff) * DIMC;
            deltaO[rg + c0] = softplusf(a0[r]);
            deltaO[rg + c1] = softplusf(a1[r]);
        }
    }
}

// ---------------------------------------------------------------------------
// Selective scan: one wave32 per (batch, channel); 2 states/lane (n, n+32).
// Bm/Cm are stored (n, n+32)-interleaved -> one b64 load per matrix per step.
__global__ void k_scan(const float* __restrict__ dlt, const float* __restrict__ xc,
                       const float* __restrict__ Bm, const float* __restrict__ Cm,
                       const float* __restrict__ Alog, const float* __restrict__ Dp,
                       float* __restrict__ y) {
    const int wv = threadIdx.x >> 5, lane = threadIdx.x & 31;
    const int sid = blockIdx.x * 8 + wv;          // 0..4095
    const int b = sid >> 8, d = sid & 255;
    const float A0 = -__expf(Alog[d * DSTATE + lane]);
    const float A1 = -__expf(Alog[d * DSTATE + lane + 32]);
    const float Dv = Dp[d];
    float h0 = 0.f, h1 = 0.f;
    size_t tix = ((size_t)b * SEQ) * DIMC + d;
    size_t bcx = ((size_t)b * SEQ) * DSTATE + lane * 2;
    for (int s = 0; s < SEQ; ++s) {
        float de = dlt[tix], xv = xc[tix];
        float dx = de * xv;
        f32x2 Bv = *(const f32x2*)(Bm + bcx);
        f32x2 Cv = *(const f32x2*)(Cm + bcx);
        h0 = h0 * __expf(de * A0) + dx * Bv[0];
        h1 = h1 * __expf(de * A1) + dx * Bv[1];
        float p = h0 * Cv[0] + h1 * Cv[1];
        for (int m = 16; m >= 1; m >>= 1) p += __shfl_xor(p, m, 32);
        if (lane == 0) y[tix] = p + Dv * xv;
        tix += DIMC; bcx += DSTATE;
    }
}

// ---------------------------------------------------------------------------
__global__ void k_combine(const f32x4* __restrict__ y1, const f32x4* __restrict__ y2,
                          const f32x4* __restrict__ z, f32x4* __restrict__ x) {
    size_t i = (size_t)blockIdx.x * 256 + threadIdx.x;
    f32x4 zv = z[i], v1 = y1[i], v2 = y2[i], xo = x[i];
    f32x4 r;
#pragma unroll
    for (int j = 0; j < 4; ++j) {
        float zz = zv[j];
        float zs = zz / (1.f + __expf(-zz));
        r[j] = (v1[j] + v2[j]) * zs + xo[j];
    }
    x[i] = r;
}

// ---------------------------------------------------------------------------
// Head part 1 (one block): pooled mean over seq (16x256), LN, then the
// 16x1000x256 classifier GEMM on the matrix cores.  Logits -> workspace.
__global__ void k_head_gemm(const float* __restrict__ xbuf,
                            const float* __restrict__ hg, const float* __restrict__ hb,
                            const _Float16* __restrict__ hw16, const float* __restrict__ hbias,
                            float* __restrict__ logits) {
    __shared__ __attribute__((aligned(16))) float    pooled[16 * DIMC];
    __shared__ __attribute__((aligned(16))) _Float16 pn16[16 * DIMC];
    const int tid = threadIdx.x, wv = tid >> 5, lane = tid & 31;
    for (int b = 0; b < BATCH; ++b) {
        const float* xb = xbuf + ((size_t)b * SEQ) * DIMC + tid;
        float s = 0.f;
        for (int i = 0; i < SEQ; ++i) s += xb[(size_t)i * DIMC];
        pooled[b * DIMC + tid] = s * (1.f / SEQ);
    }
    __syncthreads();
    const int row = (wv << 1) | (lane >> 4);
    const int sl = lane & 15;
    const float* prow = pooled + row * DIMC;
    float sum = 0.f, sq = 0.f;
#pragma unroll
    for (int i = 0; i < 16; ++i) { float v = prow[sl + (i << 4)]; sum += v; sq += v * v; }
    for (int msk = 8; msk >= 1; msk >>= 1) {
        sum += __shfl_xor(sum, msk, 32);
        sq  += __shfl_xor(sq,  msk, 32);
    }
    const float mean = sum * (1.f / DIMC);
    const float var  = sq * (1.f / DIMC) - mean * mean;
    const float rstd = rsqrtf(var + 1e-5f);
#pragma unroll
    for (int i = 0; i < 16; ++i) {
        int e = sl + (i << 4);
        pn16[row * DIMC + e] = (_Float16)((prow[e] - mean) * rstd * hg[e] + hb[e]);
    }
    __syncthreads();
    const int moff = (lane & 16) ? 8 : 0;
    for (int nt = wv; nt < 63; nt += 8) {        // 63 tiles cover 1000 cols
        int col = (nt << 4) + (lane & 15);
        int colc = col < NCLS ? col : NCLS - 1;
        float bv = hbias[colc];
        v8f acc; for (int r = 0; r < 8; ++r) acc[r] = bv;
        acc = gemm_acc(pn16, DIMC, hw16 + (size_t)colc * DIMC, 8, acc, lane);
        if (col < NCLS)
            for (int r = 0; r < 8; ++r)
                logits[(size_t)(r + moff) * NCLS + col] = acc[r];
    }
}

// Head part 2: per-batch softmax over 1000 logits.
__global__ void k_softmax(const float* __restrict__ logits, float* __restrict__ out) {
    __shared__ float red[256];
    const int b = blockIdx.x, t = threadIdx.x;
    const float* lr = logits + (size_t)b * NCLS;
    float lm = -1e30f;
    for (int o = t; o < NCLS; o += 256) lm = fmaxf(lm, lr[o]);
    red[t] = lm; __syncthreads();
    for (int st = 128; st > 0; st >>= 1) { if (t < st) red[t] = fmaxf(red[t], red[t + st]); __syncthreads(); }
    const float mx = red[0];
    __syncthreads();
    float se = 0.f;
    for (int o = t; o < NCLS; o += 256) se += __expf(lr[o] - mx);
    red[t] = se; __syncthreads();
    for (int st = 128; st > 0; st >>= 1) { if (t < st) red[t] += red[t + st]; __syncthreads(); }
    const float inv = 1.f / red[0];
    for (int o = t; o < NCLS; o += 256)
        out[(size_t)b * NCLS + o] = __expf(lr[o] - mx) * inv;
}

// ---------------------------------------------------------------------------
extern "C" void kernel_launch(void* const* d_in, const int* in_sizes, int n_in,
                              void* d_out, int out_size, void* d_ws, size_t ws_size,
                              hipStream_t stream) {
    const float* x_in     = (const float*)d_in[0];
    const float* patch_w  = (const float*)d_in[1];
    const float* patch_b  = (const float*)d_in[2];
    const float* norm_g   = (const float*)d_in[3];
    const float* norm_b   = (const float*)d_in[4];
    const float* proj_w   = (const float*)d_in[5];
    const float* proj_b   = (const float*)d_in[6];
    const float* fconv_w  = (const float*)d_in[7];
    const float* fconv_b  = (const float*)d_in[8];
    const float* bconv_w  = (const float*)d_in[9];
    const float* bconv_b  = (const float*)d_in[10];
    const float* dbc_w    = (const float*)d_in[11];
    const float* dt_w     = (const float*)d_in[12];
    const float* dt_b     = (const float*)d_in[13];
    const float* A_log    = (const float*)d_in[14];
    const float* Dp       = (const float*)d_in[15];
    const float* head_g   = (const float*)d_in[16];
    const float* head_b   = (const float*)d_in[17];
    const float* head_w   = (const float*)d_in[18];
    const float* head_bia = (const float*)d_in[19];

    // ---- workspace carve (~38 MB) ----
    char* base = (char*)d_ws;
    size_t off = 0;
    auto take = [&](size_t bytes) -> char* {
        char* r = base + off;
        off = (off + bytes + 255) & ~(size_t)255;
        return r;
    };
    const size_t MT = MTOK;
    float*     xbuf   = (float*)take(MT * DIMC * 4);
    float*     zbuf   = (float*)take(MT * DIMC * 4);
    _Float16*  z16g   = (_Float16*)take(MT * DIMC * 2);
    float*     xcf    = (float*)take(MT * DIMC * 4);
    float*     xcb    = (float*)take(MT * DIMC * 4);
    float*     dltf   = (float*)take(MT * DIMC * 4);
    float*     dltb   = (float*)take(MT * DIMC * 4);
    float*     Bmf    = (float*)take(MT * DSTATE * 4);
    float*     Cmf    = (float*)take(MT * DSTATE * 4);
    float*     Bmb    = (float*)take(MT * DSTATE * 4);
    float*     Cmb    = (float*)take(MT * DSTATE * 4);
    float*     y1     = (float*)take(MT * DIMC * 4);
    float*     y2     = (float*)take(MT * DIMC * 4);
    float*     logitW = (float*)take((size_t)BATCH * NCLS * 4);
    _Float16*  pw16   = (_Float16*)take((size_t)DIMC * 768 * 2);
    _Float16*  projW  = (_Float16*)take((size_t)DEPTHL * DIMC * DIMC * 2);
    _Float16*  fW     = (_Float16*)take((size_t)DEPTHL * DIMC * DIMC * 2);
    _Float16*  bW     = (_Float16*)take((size_t)DEPTHL * DIMC * DIMC * 2);
    _Float16*  dbcW   = (_Float16*)take((size_t)DEPTHL * EDBC * DIMC * 2);
    _Float16*  dtW    = (_Float16*)take((size_t)DEPTHL * DIMC * DTRANK * 2);
    _Float16*  hW16   = (_Float16*)take((size_t)NCLS * DIMC * 2);

    auto cvt = [&](const float* s, _Float16* d, int n) {
        k_f32_to_f16<<<(n + 255) / 256, 256, 0, stream>>>(s, d, n);
    };
    cvt(patch_w, pw16,  DIMC * 768);
    cvt(proj_w,  projW, DEPTHL * DIMC * DIMC);
    cvt(fconv_w, fW,    DEPTHL * DIMC * DIMC);
    cvt(bconv_w, bW,    DEPTHL * DIMC * DIMC);
    cvt(dbc_w,   dbcW,  DEPTHL * EDBC * DIMC);
    cvt(dt_w,    dtW,   DEPTHL * DIMC * DTRANK);
    cvt(head_w,  hW16,  NCLS * DIMC);

    k_patch<<<MT / 16, 256, 0, stream>>>(x_in, pw16, patch_b, xbuf);

    for (int l = 0; l < DEPTHL; ++l) {
        const size_t w2 = (size_t)l * DIMC * DIMC;
        k_ln_proj<<<MT / 16, 256, 0, stream>>>(
            xbuf, norm_g + l * DIMC, norm_b + l * DIMC,
            projW + w2, proj_b + l * DIMC, zbuf, z16g);
        k_conv_ssm<<<MT / 16, 256, 0, stream>>>(
            z16g, fW + w2, fconv_b + l * DIMC,
            dbcW + (size_t)l * EDBC * DIMC, dtW + (size_t)l * DIMC * DTRANK,
            dt_b + l * DIMC, xcf, Bmf, Cmf, dltf);
        k_conv_ssm<<<MT / 16, 256, 0, stream>>>(
            z16g, bW + w2, bconv_b + l * DIMC,
            dbcW + (size_t)l * EDBC * DIMC, dtW + (size_t)l * DIMC * DTRANK,
            dt_b + l * DIMC, xcb, Bmb, Cmb, dltb);
        k_scan<<<512, 256, 0, stream>>>(dltf, xcf, Bmf, Cmf,
                                        A_log + (size_t)l * DIMC * DSTATE, Dp + l * DIMC, y1);
        k_scan<<<512, 256, 0, stream>>>(dltb, xcb, Bmb, Cmb,
                                        A_log + (size_t)l * DIMC * DSTATE, Dp + l * DIMC, y2);
        k_combine<<<MT * DIMC / 1024, 256, 0, stream>>>(
            (const f32x4*)y1, (const f32x4*)y2, (const f32x4*)zbuf, (f32x4*)xbuf);
    }

    k_head_gemm<<<1, 256, 0, stream>>>(xbuf, head_g, head_b, hW16, head_bia, logitW);
    k_softmax<<<BATCH, 256, 0, stream>>>(logitW, (float*)d_out);
}
